// Net_292057776849
// MI455X (gfx1250) — matrix operand order repaired and checked by reading, not compiled, not run on hardware
//
#include <hip/hip_runtime.h>
#include <hip/hip_bf16.h>

typedef __attribute__((ext_vector_type(16))) _Float16 v16h;
typedef __attribute__((ext_vector_type(8)))  float    v8f;

#define B_   8
#define HW_  4096
#define C_   384
#define KSEL 128

#define BM 128
#define BN 128
#define BK 32
#define LDA 40   // halves per LDS row: 32 + 8 pad (keeps 16B alignment, dodges bank conflicts)
#define KSTEPS (C_ / BK)

static __device__ __forceinline__ unsigned int f2ord(float v) {
    unsigned int u = __float_as_uint(v);
    return (u & 0x80000000u) ? ~u : (u | 0x80000000u);
}

// low 32 bits of a generic LDS pointer = LDS byte offset (aperture base lives in the high half)
static __device__ __forceinline__ unsigned lds_off(const void* p) {
    return (unsigned)(reinterpret_cast<unsigned long long>(p) & 0xFFFFFFFFull);
}

// CDNA5 async DMA: global -> LDS, 16B per lane, tracked by ASYNCcnt (no VGPR round-trip)
static __device__ __forceinline__ void async_copy_b128(unsigned lds_addr, const void* gptr) {
    unsigned long long ga = reinterpret_cast<unsigned long long>(gptr);
    asm volatile("global_load_async_to_lds_b128 %0, %1, off"
                 :: "v"(lds_addr), "v"(ga) : "memory");
}

// ---------------- kernel 1: zero packed argmax keys ----------------
__global__ void zero_keys_kernel(unsigned long long* keys, int n) {
    int i = blockIdx.x * blockDim.x + threadIdx.x;
    if (i < n) keys[i] = 0ull;
}

// ---------------- kernel 2: L2-normalize + mask + f32->f16 ----------------
__global__ void prep_kernel(const float* __restrict__ f1, const float* __restrict__ f2,
                            const int* __restrict__ m1, const int* __restrict__ m2,
                            _Float16* __restrict__ o1, _Float16* __restrict__ o2) {
    int token = blockIdx.x;                  // b*hw tokens
    const float* f = blockIdx.y ? f2 : f1;
    const int*   m = blockIdx.y ? m2 : m1;
    _Float16*    o = blockIdx.y ? o2 : o1;
    size_t base = (size_t)token * C_;
    int tid = threadIdx.x;                   // 128 threads, 3 elems each
    float a0 = f[base + tid];
    float a1 = f[base + tid + 128];
    float a2 = f[base + tid + 256];
    __shared__ float red[128];
    red[tid] = a0*a0 + a1*a1 + a2*a2;
    __syncthreads();
    for (int off = 64; off > 0; off >>= 1) {
        if (tid < off) red[tid] += red[tid + off];
        __syncthreads();
    }
    float scale = (m[token] != 0) ? rsqrtf(red[0]) : 0.0f;
    o[base + tid]       = (_Float16)(a0 * scale);
    o[base + tid + 128] = (_Float16)(a1 * scale);
    o[base + tid + 256] = (_Float16)(a2 * scale);
}

// ---------------- kernel 3: per-batch mask popcount ----------------
__global__ void masksum_kernel(const int* __restrict__ m1, const int* __restrict__ m2,
                               int* __restrict__ msum) {
    int batch = blockIdx.x;
    const int* m = blockIdx.y ? m2 : m1;
    int tid = threadIdx.x;
    int s = 0;
    for (int i = tid; i < HW_; i += 256) s += (m[(size_t)batch*HW_ + i] != 0);
    __shared__ int red[256];
    red[tid] = s; __syncthreads();
    for (int off = 128; off > 0; off >>= 1) {
        if (tid < off) red[tid] += red[tid + off];
        __syncthreads();
    }
    if (tid == 0) msum[blockIdx.y * B_ + batch] = red[0];
}

// ---------------- kernel 4: WMMA GEMM (async double-buffered LDS) + fused argmax ----------------
__global__ __launch_bounds__(256, 1)
void gemm_argmax_kernel(const _Float16* __restrict__ f1h,
                        const _Float16* __restrict__ f2h,
                        const int* __restrict__ m1,
                        const int* __restrict__ m2,
                        unsigned long long* __restrict__ key1,
                        unsigned long long* __restrict__ key2) {
    __shared__ _Float16 sA[2][BM * LDA];     // double-buffered A tiles (10 KB x2)
    __shared__ _Float16 sB[2][BN * LDA];     // double-buffered B tiles
    __shared__ int sM1[BM];
    __shared__ int sM2[BN];

    const int bb = blockIdx.z;
    const int i0 = blockIdx.y * BM;
    const int j0 = blockIdx.x * BN;
    const int tid  = threadIdx.x;
    const int lane = tid & 31;
    const int wid  = tid >> 5;        // 8 waves
    const int wm = wid & 3;           // 4 wave-rows  (32 rows each)
    const int wn = wid >> 2;          // 2 wave-cols  (64 cols each)
    const int lr = lane & 15;
    const int hi = lane >> 4;

    const size_t abase = ((size_t)bb * HW_ + i0) * C_;
    const size_t bbase = ((size_t)bb * HW_ + j0) * C_;

    if (tid < 128) sM1[tid]       = m1[(size_t)bb*HW_ + i0 + tid];
    else           sM2[tid - 128] = m2[(size_t)bb*HW_ + j0 + (tid - 128)];

    // Per-thread staging coordinates: 128 rows x 32 halves per tile,
    // 256 threads x 8 halves (b128) x 2 segments, for A and B => 4 async DMAs per thread/tile.
    const int e0 = tid * 8,      r0 = e0 >> 5, c0 = e0 & 31;
    const int e1 = 2048 + tid*8, r1 = e1 >> 5, c1 = e1 & 31;
    const _Float16* gA0 = f1h + abase + (size_t)r0 * C_ + c0;
    const _Float16* gA1 = f1h + abase + (size_t)r1 * C_ + c1;
    const _Float16* gB0 = f2h + bbase + (size_t)r0 * C_ + c0;
    const _Float16* gB1 = f2h + bbase + (size_t)r1 * C_ + c1;

    auto issue_tile = [&](int kk, int buf) {
        async_copy_b128(lds_off(&sA[buf][r0 * LDA + c0]), gA0 + kk);
        async_copy_b128(lds_off(&sA[buf][r1 * LDA + c1]), gA1 + kk);
        async_copy_b128(lds_off(&sB[buf][r0 * LDA + c0]), gB0 + kk);
        async_copy_b128(lds_off(&sB[buf][r1 * LDA + c1]), gB1 + kk);
    };

    v8f zero = {};
    v8f acc[2][4];
    #pragma unroll
    for (int a = 0; a < 2; ++a)
        #pragma unroll
        for (int b = 0; b < 4; ++b) acc[a][b] = zero;

    issue_tile(0, 0);                         // prologue DMA into buffer 0

    for (int t = 0; t < KSTEPS; ++t) {
        const int cur = t & 1;
        // wave-local: all my outstanding async DMAs (tile t) have landed in LDS
        asm volatile("s_wait_asynccnt 0x0" ::: "memory");
        __syncthreads();                      // cross-wave: everyone's tile t is visible
        if (t + 1 < KSTEPS)                   // DMA-prefetch tile t+1; overlaps the WMMAs below
            issue_tile((t + 1) * BK, cur ^ 1);

        union Frag { v16h v; uint4 u[2]; };
        Frag af[2], bf[4];
        // A 16x32 f16 layout: lanes 0-15 hold K 0..7 & 16..23 of row M=lr,
        // lanes 16-31 hold K 8..15 & 24..31 (ISA 7.12.2)
        #pragma unroll
        for (int tm = 0; tm < 2; ++tm) {
            const _Float16* p = &sA[cur][(wm*32 + tm*16 + lr) * LDA + hi*8];
            af[tm].u[0] = *(const uint4*)p;
            af[tm].u[1] = *(const uint4*)(p + 16);
        }
        // B 32x16: lane = column N=lr, K hi*8..hi*8+7 & 16+hi*8..16+hi*8+7 (same order as A)
        #pragma unroll
        for (int tn = 0; tn < 4; ++tn) {
            const _Float16* p = &sB[cur][(wn*64 + tn*16 + lr) * LDA + hi*8];
            bf[tn].u[0] = *(const uint4*)p;
            bf[tn].u[1] = *(const uint4*)(p + 16);
        }
        #pragma unroll
        for (int tm = 0; tm < 2; ++tm)
            #pragma unroll
            for (int tn = 0; tn < 4; ++tn)
                acc[tm][tn] = __builtin_amdgcn_wmma_f32_16x16x32_f16(
                    false, af[tm].v, false, bf[tn].v,
                    (short)0, acc[tm][tn], false, false);

        __syncthreads();   // nobody still reads buf `cur` when iter t+1 DMAs into `cur^1`... and
                           // iter t+2 reuses `cur` only after another wait+barrier
    }

    const float NEG = -3.0e38f;

    // ---- row argmax (match1): per row best over valid j ----
    #pragma unroll
    for (int tm = 0; tm < 2; ++tm) {
        #pragma unroll
        for (int r = 0; r < 8; ++r) {
            int iloc = wm*32 + tm*16 + r + hi*8;   // C layout: VGPR r, halves split rows
            float bv = NEG; int bj = 0;
            #pragma unroll
            for (int tn = 0; tn < 4; ++tn) {
                int jloc = wn*64 + tn*16 + lr;
                float v = sM2[jloc] ? acc[tm][tn][r] : NEG;
                if (v > bv) { bv = v; bj = jloc; }   // strict > keeps smaller j on tie
            }
            #pragma unroll
            for (int msk = 1; msk <= 8; msk <<= 1) { // reduce within 16-lane half
                float ov = __shfl_xor(bv, msk, 32);
                int   oj = __shfl_xor(bj, msk, 32);
                if (ov > bv || (ov == bv && oj < bj)) { bv = ov; bj = oj; }
            }
            if (lr == 0 && bv > -1.0e38f) {
                unsigned long long key =
                    ((unsigned long long)f2ord(bv) << 32) |
                    (unsigned long long)(0xFFFFFFFFu - (unsigned)(j0 + bj));
                atomicMax(&key1[(size_t)bb*HW_ + i0 + iloc], key);
            }
        }
    }

    // ---- col argmax (match2): per column best over valid i ----
    #pragma unroll
    for (int tn = 0; tn < 4; ++tn) {
        float bv = NEG; int bi = 0;
        #pragma unroll
        for (int tm = 0; tm < 2; ++tm)
            #pragma unroll
            for (int r = 0; r < 8; ++r) {
                int iloc = wm*32 + tm*16 + r + hi*8;
                float v = sM1[iloc] ? acc[tm][tn][r] : NEG;
                if (v > bv) { bv = v; bi = iloc; }   // per-lane i ascends with (tm,r)
            }
        float ov = __shfl_xor(bv, 16, 32);           // merge the two row-halves
        int   oi = __shfl_xor(bi, 16, 32);
        if (ov > bv || (ov == bv && oi < bi)) { bv = ov; bi = oi; }
        if (hi == 0 && bv > -1.0e38f) {
            int jloc = wn*64 + tn*16 + lr;
            unsigned long long key =
                ((unsigned long long)f2ord(bv) << 32) |
                (unsigned long long)(0xFFFFFFFFu - (unsigned)(i0 + bi));
            atomicMax(&key2[(size_t)bb*HW_ + j0 + jloc], key);
        }
    }
}

// ---------------- kernel 5: packed key -> match index ----------------
__global__ void extract_kernel(const unsigned long long* __restrict__ key1,
                               const unsigned long long* __restrict__ key2,
                               const int* __restrict__ m1, const int* __restrict__ m2,
                               int* __restrict__ match1, int* __restrict__ match2) {
    int idx = blockIdx.x * blockDim.x + threadIdx.x;
    if (idx >= 2 * B_ * HW_) return;
    int dir = idx / (B_ * HW_);
    int t   = idx % (B_ * HW_);
    const unsigned long long* key = dir ? key2 : key1;
    const int* m = dir ? m2 : m1;
    int* out = dir ? match2 : match1;
    unsigned long long k = key[t];
    int match = 0;
    if (m[t] != 0 && k != 0ull)                  // masked / all-invalid rows -> 0 (argmin of 1e7 row)
        match = (int)(0xFFFFFFFFu - (unsigned)(k & 0xFFFFFFFFull));
    out[t] = match;
}

// ---------------- kernel 6: cyclic diff + stable bitonic top-K select ----------------
__global__ __launch_bounds__(1024, 1)
void select_kernel(const int* __restrict__ match1, const int* __restrict__ match2,
                   const int* __restrict__ m1, const int* __restrict__ m2,
                   const int* __restrict__ backup1, const int* __restrict__ backup2,
                   const int* __restrict__ msum, float* __restrict__ out) {
    __shared__ unsigned long long skey[HW_];     // 32 KB
    int dir   = blockIdx.x >> 3;
    int batch = blockIdx.x & 7;
    const int* mf = dir ? match2 : match1;
    const int* mb = dir ? match1 : match2;
    const int* ms = dir ? m2 : m1;
    const int* md = dir ? m1 : m2;
    const int* bk = dir ? backup2 : backup1;
    int tid = threadIdx.x;
    size_t base = (size_t)batch * HW_;

    for (int i = tid; i < HW_; i += 1024) {
        int f = mf[base + i];
        int valid = (ms[base + i] != 0) && (md[base + f] != 0);
        int cyc = mb[base + f];
        float dx = (float)(cyc >> 6) - (float)(i >> 6);   // w = 64
        float dy = (float)(cyc & 63) - (float)(i & 63);
        float val = valid ? sqrtf(dx*dx + dy*dy) : 10000.0f;
        // diff >= 0 -> float bits are order-preserving; idx in low 12 bits = stable tie-break
        skey[i] = ((unsigned long long)__float_as_uint(val) << 12) | (unsigned)i;
    }
    __syncthreads();
    for (int k = 2; k <= HW_; k <<= 1) {
        for (int j = k >> 1; j > 0; j >>= 1) {
            for (int idx = tid; idx < HW_; idx += 1024) {
                int ixj = idx ^ j;
                if (ixj > idx) {
                    unsigned long long a = skey[idx], b = skey[ixj];
                    bool up = ((idx & k) == 0);
                    if ((a > b) == up) { skey[idx] = b; skey[ixj] = a; }
                }
            }
            __syncthreads();
        }
    }
    if (tid < KSEL) {
        int choose = (int)(skey[tid] & 0xFFFull);
        if (msum[dir * B_ + batch] < KSEL)
            choose = bk[batch * KSEL + tid];
        int cm = mf[base + choose];
        float* o = out + (size_t)dir * (B_ * KSEL * 2) + ((size_t)batch * KSEL + tid) * 2;
        o[0] = (float)choose;
        o[1] = (float)cm;
    }
}

// ---------------- launcher ----------------
extern "C" void kernel_launch(void* const* d_in, const int* in_sizes, int n_in,
                              void* d_out, int out_size, void* d_ws, size_t ws_size,
                              hipStream_t stream) {
    const float* f1  = (const float*)d_in[0];
    const float* f2  = (const float*)d_in[1];
    const int*   m1  = (const int*)d_in[2];
    const int*   m2  = (const int*)d_in[3];
    const int*   bk1 = (const int*)d_in[4];
    const int*   bk2 = (const int*)d_in[5];

    char* ws = (char*)d_ws;
    size_t off = 0;
    _Float16* f1h = (_Float16*)(ws + off); off += (size_t)B_ * HW_ * C_ * 2;
    _Float16* f2h = (_Float16*)(ws + off); off += (size_t)B_ * HW_ * C_ * 2;
    unsigned long long* key1 = (unsigned long long*)(ws + off); off += (size_t)B_ * HW_ * 8;
    unsigned long long* key2 = (unsigned long long*)(ws + off); off += (size_t)B_ * HW_ * 8;
    int* match1 = (int*)(ws + off); off += (size_t)B_ * HW_ * 4;
    int* match2 = (int*)(ws + off); off += (size_t)B_ * HW_ * 4;
    int* msum   = (int*)(ws + off); off += 2 * B_ * 4;
    (void)in_sizes; (void)n_in; (void)out_size; (void)ws_size;

    {   // key1/key2 are contiguous: zero both in one sweep
        int n = 2 * B_ * HW_;
        zero_keys_kernel<<<(n + 255) / 256, 256, 0, stream>>>(key1, n);
    }
    prep_kernel<<<dim3(B_ * HW_, 2), 128, 0, stream>>>(f1, f2, m1, m2, f1h, f2h);
    masksum_kernel<<<dim3(B_, 2), 256, 0, stream>>>(m1, m2, msum);
    gemm_argmax_kernel<<<dim3(HW_ / BN, HW_ / BM, B_), 256, 0, stream>>>(
        f1h, f2h, m1, m2, key1, key2);
    {
        int n = 2 * B_ * HW_;
        extract_kernel<<<(n + 255) / 256, 256, 0, stream>>>(key1, key2, m1, m2, match1, match2);
    }
    select_kernel<<<2 * B_, 1024, 0, stream>>>(match1, match2, m1, m2, bk1, bk2,
                                               msum, (float*)d_out);
}
